// GCN_83468394431129
// MI455X (gfx1250) — matrix operand order repaired and checked by reading, not compiled, not run on hardware
//
#include <hip/hip_runtime.h>

// ---------------------------------------------------------------------------
// GCN, 3 hops on MI455X (gfx1250, wave32, WMMA).
//  - MLP runs on v_wmma_f32_16x16x32_bf16 (f32 accum).
//  - Edge gather/scatter stays L2-resident (25.6MB node features << 192MB L2).
// ---------------------------------------------------------------------------

typedef __attribute__((ext_vector_type(16))) __bf16 v16bf;
typedef __attribute__((ext_vector_type(8)))  __bf16 v8bf;
typedef __attribute__((ext_vector_type(8)))  float  v8f;
typedef __attribute__((ext_vector_type(4)))  float  v4f;

#define D_FEAT 64
#define D_HID  128
#define N_HOPS 3
#define LRELU_SLOPE 0.01f

#define W1T_LD (D_FEAT + 8)   // 72 bf16  -> 144B rows (16B aligned)
#define W2T_LD (D_HID  + 8)   // 136 bf16 -> 272B rows (16B aligned)
#define HS_LD  (D_HID  + 8)   // 136 bf16 -> 272B rows (16B aligned)
#define WAVES_PER_BLOCK 4

__device__ __forceinline__ v16bf join16(v8bf lo, v8bf hi) {
  v16bf r;
#pragma unroll
  for (int j = 0; j < 8; ++j) { r[j] = lo[j]; r[j + 8] = hi[j]; }
  return r;
}

// ---------------------------------------------------------------------------
__global__ void zero_f32(float* __restrict__ p, int n4) {
  int t = blockIdx.x * blockDim.x + threadIdx.x;
  if (t < n4) {
    v4f z = {0.f, 0.f, 0.f, 0.f};
    ((v4f*)p)[t] = z;
  }
}

__global__ void degree_kernel(const int* __restrict__ snd,
                              const int* __restrict__ rcv,
                              float* __restrict__ sdeg,
                              float* __restrict__ rdeg, int ne) {
  int t = blockIdx.x * blockDim.x + threadIdx.x;
  if (t >= ne) return;
  atomicAdd(&sdeg[snd[t]], 1.0f);   // integer-valued f32 sums: exact, order-free
  atomicAdd(&rdeg[rcv[t]], 1.0f);
}

__global__ void rsqrt_kernel(float* __restrict__ p, int n) {
  int t = blockIdx.x * blockDim.x + threadIdx.x;
  if (t < n) p[t] = rsqrtf(fmaxf(p[t], 1.0f));
}

// ---------------------------------------------------------------------------
// Fused 2-layer MLP per 16-row tile, one wave per tile.
//   y[m] = (leaky_relu((prescale[m]*x[m]) @ W1 + b1) @ W2 + b2) * postscale[m]
// ---------------------------------------------------------------------------
__launch_bounds__(WAVES_PER_BLOCK * 32)
__global__ void mlp_wmma(const float* __restrict__ x,
                         const float* __restrict__ prescale,   // may be null
                         const float* __restrict__ W1, const float* __restrict__ b1,
                         const float* __restrict__ W2, const float* __restrict__ b2,
                         const float* __restrict__ postscale,
                         float* __restrict__ y, int n) {
  __shared__ __align__(16) __bf16 W1t[D_HID][W1T_LD];   // [out col][k]
  __shared__ __align__(16) __bf16 W2t[D_FEAT][W2T_LD];  // [out col][k]
  __shared__ __align__(16) __bf16 Hs[WAVES_PER_BLOCK][16][HS_LD];
  __shared__ float b1s[D_HID];
  __shared__ float b2s[D_FEAT];

  const int t = threadIdx.x;
  // Stage weights transposed into LDS as bf16 (B-fragments become contiguous).
  for (int i = t; i < D_FEAT * D_HID; i += WAVES_PER_BLOCK * 32) {
    int k = i / D_HID, c = i % D_HID;
    W1t[c][k] = (__bf16)W1[i];
  }
  for (int i = t; i < D_HID * D_FEAT; i += WAVES_PER_BLOCK * 32) {
    int k = i / D_FEAT, c = i % D_FEAT;
    W2t[c][k] = (__bf16)W2[i];
  }
  if (t < D_HID)  b1s[t] = b1[t];
  if (t < D_FEAT) b2s[t] = b2[t];
  __syncthreads();

  const int wave = t >> 5;
  const int lane = t & 31;
  const int l16  = lane & 15;
  const int lh   = lane >> 4;                 // half-wave select
  const int tile = blockIdx.x * WAVES_PER_BLOCK + wave;
  const int row0 = tile * 16;
  if (row0 >= n) return;

  // ---- Layer 1 A-fragments (16-bit A layout: k = (j&7) + 16*(j>=8) + 8*lh)
  int m = row0 + l16;
  if (m >= n) m = n - 1;                      // clamp; stores masked later
  const float ps = prescale ? prescale[m] : 1.0f;
  const float* xr = x + (size_t)m * D_FEAT;
  v16bf a1[2];
#pragma unroll
  for (int kb = 0; kb < 2; ++kb) {
    const int c0 = kb * 32 + lh * 8;
    v4f f0 = *(const v4f*)(xr + c0);
    v4f f1 = *(const v4f*)(xr + c0 + 4);
    v4f f2 = *(const v4f*)(xr + c0 + 16);
    v4f f3 = *(const v4f*)(xr + c0 + 20);
    v16bf a;
#pragma unroll
    for (int j = 0; j < 4; ++j) {
      a[j]      = (__bf16)(f0[j] * ps);
      a[4 + j]  = (__bf16)(f1[j] * ps);
      a[8 + j]  = (__bf16)(f2[j] * ps);
      a[12 + j] = (__bf16)(f3[j] * ps);
    }
    a1[kb] = a;
  }

  // ---- Layer 1: H(16x128) = leaky_relu(A@W1 + b1), staged to LDS as bf16.
#pragma unroll
  for (int nt = 0; nt < D_HID / 16; ++nt) {
    const int col = nt * 16 + l16;
    v8f acc = {};
#pragma unroll
    for (int kb = 0; kb < 2; ++kb) {
      const __bf16* wp = &W1t[col][kb * 32 + lh * 8];
      v16bf bm = join16(*(const v8bf*)wp, *(const v8bf*)(wp + 16));
      acc = __builtin_amdgcn_wmma_f32_16x16x32_bf16(
          false, a1[kb], false, bm, (short)0, acc, false, false);
    }
    const float bb = b1s[col];
#pragma unroll
    for (int i = 0; i < 8; ++i) {            // C layout: row = i + 8*lh
      float v = acc[i] + bb;
      v = (v > 0.0f) ? v : v * LRELU_SLOPE;
      Hs[wave][i + lh * 8][col] = (__bf16)v;
    }
  }
  // Wave-private LDS tile; DS ops are in-order within a wave (DScnt).

  // ---- Layer 2 A-fragments from Hs (transpose C-layout -> A-layout via LDS)
  v16bf a2[4];
#pragma unroll
  for (int kb = 0; kb < 4; ++kb) {
    const __bf16* hp = &Hs[wave][l16][kb * 32 + lh * 8];
    a2[kb] = join16(*(const v8bf*)hp, *(const v8bf*)(hp + 16));
  }

  float prs[8];
#pragma unroll
  for (int i = 0; i < 8; ++i) {
    int gr = row0 + i + lh * 8;
    prs[i] = postscale[gr < n ? gr : n - 1];
  }

#pragma unroll
  for (int nt = 0; nt < D_FEAT / 16; ++nt) {
    const int col = nt * 16 + l16;
    v8f acc = {};
#pragma unroll
    for (int kb = 0; kb < 4; ++kb) {
      const __bf16* wp = &W2t[col][kb * 32 + lh * 8];
      v16bf bm = join16(*(const v8bf*)wp, *(const v8bf*)(wp + 16));
      acc = __builtin_amdgcn_wmma_f32_16x16x32_bf16(
          false, a2[kb], false, bm, (short)0, acc, false, false);
    }
    const float bb = b2s[col];
#pragma unroll
    for (int i = 0; i < 8; ++i) {
      const int gr = row0 + i + lh * 8;
      if (gr < n) y[(size_t)gr * D_FEAT + col] = (acc[i] + bb) * prs[i];
    }
  }
}

// ---------------------------------------------------------------------------
// Scatter: acc[rcv] += xs[snd], 16 lanes per edge, float4 per lane.
// Feature buffer (25.6MB) is L2-resident -> atomics resolve on-die.
// ---------------------------------------------------------------------------
__global__ void scatter_kernel(const float* __restrict__ xs,
                               const int* __restrict__ snd,
                               const int* __restrict__ rcv,
                               float* __restrict__ acc, int ne) {
  const int tid = blockIdx.x * blockDim.x + threadIdx.x;
  const int e = tid >> 4;
  if (e >= ne) return;
  const int l = (tid & 15) * 4;
  const int s = snd[e], r = rcv[e];
  v4f v = *(const v4f*)(xs + (size_t)s * D_FEAT + l);
  float* dst = acc + (size_t)r * D_FEAT + l;
  atomicAdd(dst + 0, v[0]);
  atomicAdd(dst + 1, v[1]);
  atomicAdd(dst + 2, v[2]);
  atomicAdd(dst + 3, v[3]);
}

__global__ void final_scale(const float* __restrict__ acc,
                            const float* __restrict__ rrs,
                            float* __restrict__ out, int n4) {
  int t = blockIdx.x * blockDim.x + threadIdx.x;
  if (t >= n4) return;                       // n4 = n_nodes * 16 (float4s)
  v4f v = ((const v4f*)acc)[t];
  float s = rrs[t >> 4];                     // 16 float4s per node row
  v4f o = {v[0] * s, v[1] * s, v[2] * s, v[3] * s};
  ((v4f*)out)[t] = o;
}

// ---------------------------------------------------------------------------
extern "C" void kernel_launch(void* const* d_in, const int* in_sizes, int n_in,
                              void* d_out, int out_size, void* d_ws, size_t ws_size,
                              hipStream_t stream) {
  const float* nodes = (const float*)d_in[0];
  const int*   snd   = (const int*)d_in[1];
  const int*   rcv   = (const int*)d_in[2];
  const float* W1    = (const float*)d_in[3];
  const float* b1    = (const float*)d_in[4];
  const float* W2    = (const float*)d_in[5];
  const float* b2    = (const float*)d_in[6];

  const int n  = in_sizes[0] / D_FEAT;   // 100000 nodes
  const int ne = in_sizes[1];            // 1600000 edges

  float* srs = (float*)d_ws;             // send-degree -> rsqrt  [n]
  float* rrs = srs + n;                  // recv-degree -> rsqrt  [n]
  float* xs  = rrs + n;                  // MLP output / gather src [n*64]
  float* acc = xs + (size_t)n * D_FEAT;  // scatter accumulator     [n*64]

  const int B = 256;
  // Degrees (hop-invariant): zero, histogram, rsqrt(max(deg,1)).
  zero_f32<<<((2 * n / 4) + B - 1) / B, B, 0, stream>>>(srs, 2 * n / 4);
  degree_kernel<<<(ne + B - 1) / B, B, 0, stream>>>(snd, rcv, srs, rrs, ne);
  rsqrt_kernel<<<(2 * n + B - 1) / B, B, 0, stream>>>(srs, 2 * n);

  const int tiles = (n + 15) / 16;
  const int mlp_blocks = (tiles + WAVES_PER_BLOCK - 1) / WAVES_PER_BLOCK;
  const int acc4 = n * (D_FEAT / 4);

  for (int hop = 0; hop < N_HOPS; ++hop) {
    const float* xin = (hop == 0) ? nodes : acc;
    const float* pre = (hop == 0) ? nullptr : rrs;   // fuse prev hop's recv scale
    mlp_wmma<<<mlp_blocks, WAVES_PER_BLOCK * 32, 0, stream>>>(
        xin, pre,
        W1 + (size_t)hop * D_FEAT * D_HID, b1 + (size_t)hop * D_HID,
        W2 + (size_t)hop * D_HID * D_FEAT, b2 + (size_t)hop * D_FEAT,
        srs, xs, n);
    zero_f32<<<(acc4 + B - 1) / B, B, 0, stream>>>(acc, acc4);
    scatter_kernel<<<((ne * 16) + B - 1) / B, B, 0, stream>>>(xs, snd, rcv, acc, ne);
  }
  final_scale<<<((n * 16) + B - 1) / B, B, 0, stream>>>(acc, rrs, (float*)d_out, n * 16);
}
